// LocalAttention_21251498180794
// MI455X (gfx1250) — matrix operand order repaired and checked by reading, not compiled, not run on hardware
//
#include <hip/hip_runtime.h>

#define CH   64
#define IMW  128
#define IMH  128
#define HW   (IMW*IMH)
#define NB   4
#define NPT  128      // pixels per GEMM tile
#define TS   16       // attention tile side
#define HALO 20       // TS + 2*pad

typedef __attribute__((ext_vector_type(16))) __bf16 v16bf;
typedef __attribute__((ext_vector_type(8)))  float  v8f;
typedef __attribute__((ext_vector_type(4)))  unsigned int u32x4;
typedef __attribute__((ext_vector_type(8)))  int   i32x8;
typedef __attribute__((ext_vector_type(4)))  int   i32x4;

__device__ __forceinline__ unsigned short f2bf(float f) {
    union { float f; unsigned u; } x; x.f = f;
    unsigned u = x.u + 0x7FFFu + ((x.u >> 16) & 1u);   // round-to-nearest-even
    return (unsigned short)(u >> 16);
}
__device__ __forceinline__ float bf2f(unsigned short h) {
    union { unsigned u; float f; } x; x.u = ((unsigned)h) << 16;
    return x.f;
}

// K index inside a 16-bit A fragment (16x32): VGPR-pair r, lane half, sub-elem s
__device__ __forceinline__ int afrag_k(int r, int half, int s) {
    int base = (r < 4) ? (2 * r) : (16 + 2 * (r - 4));
    return base + 8 * half + s;
}

// ---------------------------------------------------------------------------
// TDM: DMA a 2-D tile (tile_d0 x tile_d1 elems, row stride stride0 elems)
// from global memory into LDS at byte offset lds_off (compact row-major).
// D# layout per CDNA5 ISA 8.3/8.4. Issue from ONE wave only (EXEC ignored).
// ---------------------------------------------------------------------------
__device__ __forceinline__ void tdm_load_2d(unsigned lds_off, const void* gaddr,
                                            int ds_log2,       // 0:1B 1:2B 2:4B
                                            unsigned tile_d0, unsigned tile_d1,
                                            unsigned long long tensor_d0,
                                            unsigned long long tensor_d1,
                                            unsigned long long stride0)
{
    unsigned long long ga = (unsigned long long)gaddr;
    u32x4 g0;
    g0[0] = 1u;                                              // count=1, no gather
    g0[1] = lds_off;                                         // LDS byte address
    g0[2] = (unsigned)(ga & 0xFFFFFFFFu);                    // global_addr lo
    g0[3] = (unsigned)((ga >> 32) & 0x01FFFFFFu) | (2u << 30); // addr hi | type=2
    i32x8 g1;
    g1[0] = (int)((unsigned)ds_log2 << 16);                  // mask=0, data_size
    g1[1] = (int)((tensor_d0 & 0xFFFFu) << 16);              // tensor_dim0[15:0]
    g1[2] = (int)(((tensor_d0 >> 16) & 0xFFFFu) |
                  ((tensor_d1 & 0xFFFFu) << 16));            // td0 hi | td1 lo
    g1[3] = (int)(((tensor_d1 >> 16) & 0xFFFFu) |
                  ((tile_d0 & 0xFFFFu) << 16));              // td1 hi | tile_dim0
    g1[4] = (int)(tile_d1 & 0xFFFFu);                        // tile_dim1, tile_dim2=0
    g1[5] = (int)(stride0 & 0xFFFFFFFFu);                    // dim0 stride lo
    g1[6] = (int)((stride0 >> 32) & 0xFFFFu);                // stride hi, td1_stride=0
    g1[7] = 0;
    i32x4 z4 = {0, 0, 0, 0};
#if defined(__clang_major__) && (__clang_major__ >= 23)
    i32x8 z8 = {0, 0, 0, 0, 0, 0, 0, 0};
    __builtin_amdgcn_tensor_load_to_lds(g0, g1, z4, z4, z8, 0);
#else
    __builtin_amdgcn_tensor_load_to_lds(g0, g1, z4, z4, 0);
#endif
}

__device__ __forceinline__ unsigned lds_off32(const void* p) {
    // generic LDS addr: addr[31:0] is the LDS byte offset (flat aperture map)
    return (unsigned)(unsigned long long)p;
}

// ---------------------------------------------------------------------------
// Kernel 1: fused q = wy@y + by  and  v = wx@x + bx   (bf16 out, (pixel,ch))
// dynamic LDS: [0,32K) xraw f32 | [32K,48K) wraw f32 | [48K,56K) A frags
//              [56K,72K) B frags | [72K,88K) out tile | [88K..) bias
// ---------------------------------------------------------------------------
__global__ void __launch_bounds__(256)
qv_gemm(const float* __restrict__ x,  const float* __restrict__ y,
        const float* __restrict__ wx, const float* __restrict__ bx,
        const float* __restrict__ wy, const float* __restrict__ by,
        unsigned short* __restrict__ q_ws, unsigned short* __restrict__ v_ws)
{
    extern __shared__ char smem[];
    float*          xraw   = (float*)(smem);
    float*          wraw   = (float*)(smem + 32768);
    unsigned short* lds_a  = (unsigned short*)(smem + 49152);
    unsigned short* lds_b  = (unsigned short*)(smem + 57344);
    unsigned short* lds_o  = (unsigned short*)(smem + 73728);
    float*          lds_bs = (float*)(smem + 90112);

    const int t    = threadIdx.x;
    const bool isQ = blockIdx.x < 512;
    const int tile = blockIdx.x & 511;
    const int bb   = tile >> 7;
    const int p0   = (tile & 127) * NPT;

    const float* Xin  = (isQ ? y : x) + (size_t)bb * CH * HW;
    const float* W    = isQ ? wy : wx;
    const float* bias = isQ ? by : bx;
    unsigned short* Out = (isQ ? q_ws : v_ws) + ((size_t)bb * HW + p0) * CH;

    // ---- TDM: DMA weight matrix + input tile into LDS (wave 0 only)
    if (t < 32) {
        tdm_load_2d(lds_off32(wraw), W,        2, CH,  CH, CH,    CH, CH);
        tdm_load_2d(lds_off32(xraw), Xin + p0, 2, NPT, CH, HW,    CH, HW);
        __builtin_amdgcn_s_wait_tensorcnt(0);
    }
    if (t < CH) lds_bs[t] = bias[t];
    __syncthreads();

    // ---- swizzle weights into A-fragment order: thread t == slot(mt,kt,lane)
    {
        const int mt = t >> 6, kt = (t >> 5) & 1, lane = t & 31;
        const int m = mt * 16 + (lane & 15), half = lane >> 4;
        unsigned short* dst = &lds_a[t * 16];
#pragma unroll
        for (int e = 0; e < 16; ++e) {
            int r = e >> 1, s = e & 1;
            int k = kt * 32 + afrag_k(r, half, s);
            dst[e] = f2bf(wraw[m * CH + k]);
        }
    }
    // ---- swizzle input tile into B-fragment order
#pragma unroll
    for (int i = 0; i < 32; ++i) {
        int flat = i * 256 + t;            // 64 ch * 128 px
        int c = flat >> 7, p = flat & 127;
        float vv = xraw[c * NPT + p];
        int kt = c >> 5, lane = ((c >> 3) & 1) * 16 + (p & 15), nt = p >> 4, e = (c & 7) + ((c >> 4) & 1) * 8;
        lds_b[((kt * 8 + nt) * 32 + lane) * 16 + e] = f2bf(vv);
    }
    __syncthreads();

    const int wv   = t >> 5;     // wave id == N-subtile
    const int lane = t & 31;
    const int half = lane >> 4;

    v16bf b0 = *(const v16bf*)&lds_b[((0 * 8 + wv) * 32 + lane) * 16];
    v16bf b1 = *(const v16bf*)&lds_b[((1 * 8 + wv) * 32 + lane) * 16];

#pragma unroll
    for (int mt = 0; mt < 4; ++mt) {
        v16bf a0 = *(const v16bf*)&lds_a[((mt * 2 + 0) * 32 + lane) * 16];
        v16bf a1 = *(const v16bf*)&lds_a[((mt * 2 + 1) * 32 + lane) * 16];
        v8f acc;
#pragma unroll
        for (int v = 0; v < 8; ++v) acc[v] = lds_bs[mt * 16 + v + 8 * half];
        acc = __builtin_amdgcn_wmma_f32_16x16x32_bf16(false, a0, false, b0,
                                                      (short)0, acc, false, false);
        acc = __builtin_amdgcn_wmma_f32_16x16x32_bf16(false, a1, false, b1,
                                                      (short)0, acc, false, false);
        int p = wv * 16 + (lane & 15);
#pragma unroll
        for (int v = 0; v < 8; v += 2) {
            unsigned pk = (unsigned)f2bf(acc[v]) | ((unsigned)f2bf(acc[v + 1]) << 16);
            *(unsigned*)&lds_o[p * CH + mt * 16 + v + 8 * half] = pk;
        }
    }
    __syncthreads();

    // ---- coalesced copy-out (16 KB as uint4)
    const uint4* src = (const uint4*)lds_o;
    uint4*       dst = (uint4*)Out;
#pragma unroll
    for (int i = 0; i < 4; ++i) dst[i * 256 + t] = src[i * 256 + t];
}

// ---------------------------------------------------------------------------
// Kernel 2: local attention: scores -> softmax(25) -> aggregate (bf16 out)
// ---------------------------------------------------------------------------
__global__ void __launch_bounds__(256)
local_attn(const unsigned short* __restrict__ q_ws,
           const unsigned short* __restrict__ v_ws,
           unsigned short* __restrict__ attn_ws)
{
    extern __shared__ unsigned short smem_a[];
    unsigned short* lds_q = smem_a;                       // 20*20*64 bf16
    unsigned short* lds_v = smem_a + HALO * HALO * CH;

    const int t   = threadIdx.x;
    const int bid = blockIdx.x;                           // 4 * 8 * 8
    const int bb  = bid >> 6;
    const int th  = (bid >> 3) & 7, tw = bid & 7;
    const int h0  = th * TS, w0 = tw * TS;

    // ---- stage zero-padded halos (uint == 2 channels per access)
    for (int i = 0; i < 50; ++i) {
        int flat = i * 256 + t;                           // 400 px * 32 ch-pairs
        int hp = flat >> 5, c2 = flat & 31;
        int hr = hp / HALO, hc = hp % HALO;
        int gh = h0 - 2 + hr, gw = w0 - 2 + hc;
        unsigned qv = 0u, vv = 0u;
        if (gh >= 0 && gh < IMH && gw >= 0 && gw < IMW) {
            size_t gp = ((size_t)bb * HW + gh * IMW + gw) * CH;
            qv = *(const unsigned*)&q_ws[gp + c2 * 2];
            vv = *(const unsigned*)&v_ws[gp + c2 * 2];
        }
        *(unsigned*)&lds_q[hp * CH + c2 * 2] = qv;
        *(unsigned*)&lds_v[hp * CH + c2 * 2] = vv;
    }
    __syncthreads();

    const int pr = t >> 4, pc = t & 15;
    const int chp = (pr + 2) * HALO + (pc + 2);

    float qc[CH];
#pragma unroll
    for (int c2 = 0; c2 < 32; ++c2) {
        unsigned u = *(const unsigned*)&lds_q[chp * CH + c2 * 2];
        qc[c2 * 2]     = bf2f((unsigned short)(u & 0xFFFFu));
        qc[c2 * 2 + 1] = bf2f((unsigned short)(u >> 16));
    }

    float sc[25];
#pragma unroll
    for (int dy = 0; dy < 5; ++dy)
#pragma unroll
        for (int dx = 0; dx < 5; ++dx) {
            int np = (pr + dy) * HALO + (pc + dx);
            const unsigned short* qn = &lds_q[np * CH];
            float s = 0.f;
#pragma unroll
            for (int c2 = 0; c2 < 32; ++c2) {
                unsigned u = *(const unsigned*)&qn[c2 * 2];
                s += qc[c2 * 2]     * bf2f((unsigned short)(u & 0xFFFFu));
                s += qc[c2 * 2 + 1] * bf2f((unsigned short)(u >> 16));
            }
            sc[dy * 5 + dx] = s;
        }

    float m = sc[0];
#pragma unroll
    for (int i = 1; i < 25; ++i) m = fmaxf(m, sc[i]);
    float sum = 0.f;
#pragma unroll
    for (int i = 0; i < 25; ++i) { sc[i] = __expf(sc[i] - m); sum += sc[i]; }
    float inv = 1.f / sum;

    float oa[CH];
#pragma unroll
    for (int c = 0; c < CH; ++c) oa[c] = 0.f;
#pragma unroll
    for (int dy = 0; dy < 5; ++dy)
#pragma unroll
        for (int dx = 0; dx < 5; ++dx) {
            float a = sc[dy * 5 + dx] * inv;
            int np = (pr + dy) * HALO + (pc + dx);
            const unsigned short* vn = &lds_v[np * CH];
#pragma unroll
            for (int c2 = 0; c2 < 32; ++c2) {
                unsigned u = *(const unsigned*)&vn[c2 * 2];
                oa[c2 * 2]     += a * bf2f((unsigned short)(u & 0xFFFFu));
                oa[c2 * 2 + 1] += a * bf2f((unsigned short)(u >> 16));
            }
        }

    unsigned short* out = &attn_ws[((size_t)bb * HW + (h0 + pr) * IMW + (w0 + pc)) * CH];
#pragma unroll
    for (int c2 = 0; c2 < 32; ++c2) {
        unsigned pk = (unsigned)f2bf(oa[c2 * 2]) | ((unsigned)f2bf(oa[c2 * 2 + 1]) << 16);
        *(unsigned*)&out[c2 * 2] = pk;
    }
}

// ---------------------------------------------------------------------------
// Kernel 3: out = wo @ attn + bo + x   (f32 out, (b,c,h,w))
// dynamic LDS: [0,16K) braw bf16 | [16K,32K) wraw f32 | [32K,40K) A frags
//              [40K,56K) B frags | [56K,88K) f32 out tile | [88K..) bias
// ---------------------------------------------------------------------------
__global__ void __launch_bounds__(256)
out_gemm(const unsigned short* __restrict__ attn_ws,
         const float* __restrict__ wo, const float* __restrict__ bo,
         const float* __restrict__ x, float* __restrict__ out)
{
    extern __shared__ char smem[];
    unsigned short* braw   = (unsigned short*)(smem);
    float*          wraw   = (float*)(smem + 16384);
    unsigned short* lds_a  = (unsigned short*)(smem + 32768);
    unsigned short* lds_b  = (unsigned short*)(smem + 40960);
    float*          lds_of = (float*)(smem + 57344);
    float*          lds_bs = (float*)(smem + 90112);

    const int t    = threadIdx.x;
    const int tile = blockIdx.x;                              // 512
    const int bb   = tile >> 7;
    const int p0   = (tile & 127) * NPT;

    const unsigned short* Bin = attn_ws + ((size_t)bb * HW + p0) * CH;
    const float* xb = x + (size_t)bb * CH * HW;

    // ---- TDM: DMA weights + bf16 attention tile into LDS (wave 0 only)
    if (t < 32) {
        tdm_load_2d(lds_off32(wraw), wo,  2, CH, CH,  CH, CH,          CH);
        tdm_load_2d(lds_off32(braw), Bin, 1, CH, NPT, CH, (size_t)NB * HW, CH);
        __builtin_amdgcn_s_wait_tensorcnt(0);
    }
    if (t < CH) lds_bs[t] = bo[t];

    // ---- prefetch residual x tile (64 rows x 512B) while GEMM runs
    __builtin_prefetch(&xb[(size_t)(t >> 2) * HW + p0 + (t & 3) * 32], 0, 0);
    __syncthreads();

    // ---- swizzle weights into A-fragment order
    {
        const int mt = t >> 6, kt = (t >> 5) & 1, lane = t & 31;
        const int m = mt * 16 + (lane & 15), half = lane >> 4;
        unsigned short* dst = &lds_a[t * 16];
#pragma unroll
        for (int e = 0; e < 16; ++e) {
            int r = e >> 1, s = e & 1;
            int k = kt * 32 + afrag_k(r, half, s);
            dst[e] = f2bf(wraw[m * CH + k]);
        }
    }
    // ---- swizzle bf16 tile into B-fragment order
#pragma unroll
    for (int i = 0; i < 32; ++i) {
        int flat = i * 256 + t;
        int p = flat >> 6, c = flat & 63;
        unsigned short vv = braw[p * CH + c];
        int kt = c >> 5, lane = ((c >> 3) & 1) * 16 + (p & 15), nt = p >> 4, e = (c & 7) + ((c >> 4) & 1) * 8;
        lds_b[((kt * 8 + nt) * 32 + lane) * 16 + e] = vv;
    }
    __syncthreads();

    const int wv   = t >> 5;
    const int lane = t & 31;
    const int half = lane >> 4;

    v16bf b0 = *(const v16bf*)&lds_b[((0 * 8 + wv) * 32 + lane) * 16];
    v16bf b1 = *(const v16bf*)&lds_b[((1 * 8 + wv) * 32 + lane) * 16];

#pragma unroll
    for (int mt = 0; mt < 4; ++mt) {
        v16bf a0 = *(const v16bf*)&lds_a[((mt * 2 + 0) * 32 + lane) * 16];
        v16bf a1 = *(const v16bf*)&lds_a[((mt * 2 + 1) * 32 + lane) * 16];
        v8f acc;
#pragma unroll
        for (int v = 0; v < 8; ++v) acc[v] = lds_bs[mt * 16 + v + 8 * half];
        acc = __builtin_amdgcn_wmma_f32_16x16x32_bf16(false, a0, false, b0,
                                                      (short)0, acc, false, false);
        acc = __builtin_amdgcn_wmma_f32_16x16x32_bf16(false, a1, false, b1,
                                                      (short)0, acc, false, false);
        int p = wv * 16 + (lane & 15);
#pragma unroll
        for (int v = 0; v < 8; ++v)
            lds_of[(mt * 16 + v + 8 * half) * NPT + p] = acc[v];
    }
    __syncthreads();

    float* ob = out + (size_t)bb * CH * HW;
#pragma unroll
    for (int i = 0; i < 32; ++i) {
        int flat = i * 256 + t;
        int c = flat >> 7, p = flat & 127;
        size_t gi = (size_t)c * HW + p0 + p;
        ob[gi] = lds_of[c * NPT + p] + xb[gi];
    }
}

// ---------------------------------------------------------------------------
extern "C" void kernel_launch(void* const* d_in, const int* in_sizes, int n_in,
                              void* d_out, int out_size, void* d_ws, size_t ws_size,
                              hipStream_t stream)
{
    (void)in_sizes; (void)n_in; (void)out_size; (void)ws_size;
    const float* x  = (const float*)d_in[0];
    const float* y  = (const float*)d_in[1];
    const float* wx = (const float*)d_in[2];
    const float* bx = (const float*)d_in[3];
    const float* wy = (const float*)d_in[4];
    const float* by = (const float*)d_in[5];
    const float* wo = (const float*)d_in[6];
    const float* bo = (const float*)d_in[7];
    float* out = (float*)d_out;

    const size_t tsz = (size_t)NB * HW * CH * sizeof(unsigned short);  // 8 MB
    unsigned short* q_ws = (unsigned short*)d_ws;
    unsigned short* v_ws = (unsigned short*)((char*)d_ws + tsz);
    unsigned short* a_ws = (unsigned short*)((char*)d_ws + 2 * tsz);

    const int gemm_lds = 90368;   // see layout comments
    qv_gemm<<<dim3(1024), dim3(256), gemm_lds, stream>>>(x, y, wx, bx, wy, by, q_ws, v_ws);

    const int attn_lds = 2 * HALO * HALO * CH * (int)sizeof(unsigned short); // 100 KB
    local_attn<<<dim3(256), dim3(256), attn_lds, stream>>>(q_ws, v_ws, a_ws);

    out_gemm<<<dim3(512), dim3(256), gemm_lds, stream>>>(a_ws, wo, bo, x, out);
}
